// NeuralODE_27410481283716
// MI455X (gfx1250) — matrix operand order repaired and checked by reading, not compiled, not run on hardware
//
#include <hip/hip_runtime.h>
#include <hip/hip_bf16.h>

// Neural ODE  dx/dt = tanh(x @ W1^T + b1) @ W2^T + b2, fixed-step RK4.
// B=8192, D=64, H=256, T=50. One wave per workgroup owns 16 batch rows for
// the whole trajectory. Both GEMMs use v_wmma_f32_16x16x32_f16.
//
// Round 3: weight fragments are fully REGISTER-RESIDENT (~672 VGPRs; gfx1250
// wave32 allows 1024 with MSB addressing). Per-eval LDS traffic collapses to
// 3 X-fragment loads + 32 stage stores; inner loop is WMMA + native v_tanh_f32
// (TRANS, co-executes with WMMA). Biases folded into the GEMMs by K-extension.

typedef _Float16 v8h  __attribute__((ext_vector_type(8)));
typedef _Float16 v16h __attribute__((ext_vector_type(16)));
typedef float    v8f  __attribute__((ext_vector_type(8)));

#define BSZ   8192
#define DD    64
#define HH    256
#define TT    50
#define NSUB  4      // RK4 substeps per output interval
#define BM    16     // batch rows per workgroup (one wave)
#define LK1   96     // D=64 + ones-column(64) + zero pad -> K multiple of 32
#define LK2   288    // H=256 + bias column(256) + zero pad

#define WMMA_F16(A, B, C) \
  __builtin_amdgcn_wmma_f32_16x16x32_f16(false, (A), false, (B), (short)0, (C), false, false)

__device__ __forceinline__ float fast_tanh(float x) {
#if __has_builtin(__builtin_amdgcn_tanhf)
  return __builtin_amdgcn_tanhf(x);
#else
  float y;
  asm("v_tanh_f32 %0, %1" : "=v"(y) : "v"(x));
  return y;
#endif
}

// ---- fragment loaders (per-lane LDS addressing per CDNA5 7.12.2 layouts) ----
// A-matrix 16x32 f16 from S[m][k] (k contiguous, ld=ldk):
//   lane l(<16): M=l, K {kb+0..7},{kb+16..23}; lane l+16: {kb+8..15},{kb+24..31}
__device__ __forceinline__ v16h ldsA(const _Float16* S, int ldk, int m, int kb, int hf) {
  v16h a;
  ((v8h*)&a)[0] = *(const v8h*)(S + m * ldk + kb + hf * 8);
  ((v8h*)&a)[1] = *(const v8h*)(S + m * ldk + kb + 16 + hf * 8);
  return a;
}
// B-matrix 32x16 f16 from W[n][k] (k contiguous): lane l: N=l, K {kb+hf*8..+7},{kb+16+hf*8..+7}
__device__ __forceinline__ v16h ldsB(const _Float16* W, int ldk, int n, int kb, int hf) {
  v16h b;
  const v8h* p = (const v8h*)(W + n * ldk + kb + hf * 8);
  ((v8h*)&b)[0] = p[0];
  ((v8h*)&b)[1] = p[2];
  return b;
}

// one derivative evaluation: k = tanh(Xs @ W1ext^T) @ W2ext^T (biases folded in)
// w1f: register-resident A-fragments of W1ext   [mtile][kstep]
// w2f: register-resident B-fragments of W2ext   [ntile][kstep] (kstep 8 = b2)
__device__ __forceinline__ void fdot_eval(const _Float16* Xs,
                                          const v16h (&w1f)[16][3],
                                          const v16h (&w2f)[4][9],
                                          v16h ones_frag, int l, int hf,
                                          float* kout) {
  // X^T as B-matrix: 3 fragments reused by all 16 hidden M-tiles
  v16h xb[3];
#pragma unroll
  for (int j = 0; j < 3; ++j) xb[j] = ldsB(Xs, LK1, l, j * 32, hf);

  v8f acc2[4];
#pragma unroll
  for (int nt = 0; nt < 4; ++nt) acc2[nt] = (v8f){0, 0, 0, 0, 0, 0, 0, 0};

#pragma unroll
  for (int kk = 0; kk < 8; ++kk) {
    // layer 1 (transposed): hidden tiles mt=2kk, 2kk+1 for 16 batch cols
    v8f h0 = {0, 0, 0, 0, 0, 0, 0, 0};
    v8f h1 = {0, 0, 0, 0, 0, 0, 0, 0};
#pragma unroll
    for (int j = 0; j < 3; ++j) {
      h0 = WMMA_F16(w1f[2 * kk][j], xb[j], h0);
      h1 = WMMA_F16(w1f[2 * kk + 1][j], xb[j], h1);
    }
    // tanh + pack: C/D regs of tiles (2kk,2kk+1) ARE the matmul2 A-fragment
    // for K-base 32*kk (lane = batch row, K = hidden index).
    v16h hfrag;
#pragma unroll
    for (int j = 0; j < 8; ++j) {
      hfrag[j]     = (_Float16)fast_tanh(h0[j]);
      hfrag[j + 8] = (_Float16)fast_tanh(h1[j]);
    }
    // layer 2 partial: dX += H[:,32kk:32kk+32] @ W2^T[32kk:32kk+32,:]
#pragma unroll
    for (int nt = 0; nt < 4; ++nt) acc2[nt] = WMMA_F16(hfrag, w2f[nt][kk], acc2[nt]);
  }
  // bias step: ones A-fragment x W2ext column block 256 (holds b2)
#pragma unroll
  for (int nt = 0; nt < 4; ++nt) {
    acc2[nt] = WMMA_F16(ones_frag, w2f[nt][8], acc2[nt]);
#pragma unroll
    for (int r = 0; r < 8; ++r) kout[nt * 8 + r] = acc2[nt][r];
  }
}

// stage eval point x + c*k into Xs[batchrow][0..63] (cols 64..95 preset once)
__device__ __forceinline__ void stageX(_Float16* Xs, const float* x, const float* k,
                                       float c, int l, int hf) {
#pragma unroll
  for (int nt = 0; nt < 4; ++nt)
#pragma unroll
    for (int r = 0; r < 8; ++r)
      Xs[(r + 8 * hf) * LK1 + nt * 16 + l] =
          (_Float16)(x[nt * 8 + r] + c * k[nt * 8 + r]);
}

__global__ __launch_bounds__(32) void node_rk4_kernel(
    const float* __restrict__ x0, const float* __restrict__ tg,
    const float* __restrict__ W1, const float* __restrict__ b1,
    const float* __restrict__ W2, const float* __restrict__ b2,
    float* __restrict__ out) {
  __shared__ __align__(16) _Float16 W1s[HH * LK1];  // staging only
  __shared__ __align__(16) _Float16 W2s[DD * LK2];  // staging only
  __shared__ __align__(16) _Float16 Xs[BM * LK1];   // [batchrow][k]; col64=1

  const int tid = threadIdx.x;
  const int l = tid & 15;
  const int hf = (tid >> 4) & 1;

  // stage extended weights once (f16) into LDS, then lift into registers
  for (int i = tid; i < HH * LK1; i += 32) {
    int h = i / LK1, k = i % LK1;
    float v = (k < DD) ? W1[h * DD + k] : ((k == DD) ? b1[h] : 0.0f);
    W1s[i] = (_Float16)v;
  }
  for (int i = tid; i < DD * LK2; i += 32) {
    int d = i / LK2, k = i % LK2;
    float v = (k < HH) ? W2[d * HH + k] : ((k == HH) ? b2[d] : 0.0f);
    W2s[i] = (_Float16)v;
  }
  // preset ones-column / zero padding of X stage (never overwritten)
  for (int i = tid; i < BM * LK1; i += 32) {
    int k = i % LK1;
    if (k >= DD) Xs[i] = (k == DD) ? (_Float16)1.0f : (_Float16)0.0f;
  }
  __syncthreads();

  // lift all weight fragments into registers (~672 VGPRs, live for the
  // whole trajectory; gfx1250 wave32 supports 1024 VGPRs via MSB addressing)
  v16h w1f[16][3];
#pragma unroll
  for (int mt = 0; mt < 16; ++mt)
#pragma unroll
    for (int j = 0; j < 3; ++j) w1f[mt][j] = ldsA(W1s, LK1, mt * 16 + l, j * 32, hf);
  v16h w2f[4][9];
#pragma unroll
  for (int nt = 0; nt < 4; ++nt)
#pragma unroll
    for (int kk = 0; kk < 9; ++kk)
      w2f[nt][kk] = ldsB(W2s, LK2, nt * 16 + l, kk * 32, hf);

  // constant A-fragment for the b2 step: K==256 -> element 0 of hf==0 lanes
  v16h ones_frag = {};
  ones_frag[0] = (_Float16)(hf ? 0.0f : 1.0f);

  const int gr0 = blockIdx.x * BM;  // global batch-row base of this wave's tile

  // x in WMMA C/D layout of matmul2: x[nt*8+r] = X[row r+8*hf][col nt*16+l]
  float x[32];
#pragma unroll
  for (int nt = 0; nt < 4; ++nt)
#pragma unroll
    for (int r = 0; r < 8; ++r)
      x[nt * 8 + r] = x0[(size_t)(gr0 + r + 8 * hf) * DD + nt * 16 + l];

  // t=0 snapshot
#pragma unroll
  for (int nt = 0; nt < 4; ++nt)
#pragma unroll
    for (int r = 0; r < 8; ++r)
      out[(size_t)(gr0 + r + 8 * hf) * (TT * DD) + nt * 16 + l] = x[nt * 8 + r];

  float tprev = tg[0];
  for (int ti = 1; ti < TT; ++ti) {
    float tcur = tg[ti];
    float hstep = (tcur - tprev) * (1.0f / NSUB);
    tprev = tcur;
    for (int s = 0; s < NSUB; ++s) {
      const float ac[4] = {0.0f, 0.5f, 0.5f, 1.0f};
      const float wc[4] = {1.0f, 2.0f, 2.0f, 1.0f};
      float k[32], ksum[32];
#pragma unroll
      for (int i = 0; i < 32; ++i) { k[i] = 0.0f; ksum[i] = 0.0f; }
#pragma unroll 1  // single fdot instantiation keeps I$ footprint small
      for (int e = 0; e < 4; ++e) {
        stageX(Xs, x, k, ac[e] * hstep, l, hf);
        __syncthreads();  // order Xs stores before fragment loads (1-wave WG: cheap)
        fdot_eval(Xs, w1f, w2f, ones_frag, l, hf, k);
#pragma unroll
        for (int i = 0; i < 32; ++i) ksum[i] += wc[e] * k[i];
      }
      const float h6 = hstep * (1.0f / 6.0f);
#pragma unroll
      for (int i = 0; i < 32; ++i) x[i] += h6 * ksum[i];
    }
    // snapshot at time ti: out[b][ti][d]
#pragma unroll
    for (int nt = 0; nt < 4; ++nt)
#pragma unroll
      for (int r = 0; r < 8; ++r)
        out[(size_t)(gr0 + r + 8 * hf) * (TT * DD) + (size_t)ti * DD + nt * 16 + l] =
            x[nt * 8 + r];
  }
}

extern "C" void kernel_launch(void* const* d_in, const int* in_sizes, int n_in,
                              void* d_out, int out_size, void* d_ws, size_t ws_size,
                              hipStream_t stream) {
  const float* x0 = (const float*)d_in[0];
  const float* t  = (const float*)d_in[1];
  const float* W1 = (const float*)d_in[2];
  const float* b1 = (const float*)d_in[3];
  const float* W2 = (const float*)d_in[4];
  const float* b2 = (const float*)d_in[5];
  float* out = (float*)d_out;

  dim3 grid(BSZ / BM);  // 512 one-wave workgroups
  dim3 block(32);
  node_rk4_kernel<<<grid, block, 0, stream>>>(x0, t, W1, b1, W2, b2, out);
}
